// Attention_5892695130166
// MI455X (gfx1250) — matrix operand order repaired and checked by reading, not compiled, not run on hardware
//
#include <hip/hip_runtime.h>
#include <hip/hip_bf16.h>

// ---------------- constants ----------------
#define S_LEN  2048
#define HID    1024
#define NHEAD  16
#define DEPTH  64
#define H3     3072

typedef __attribute__((ext_vector_type(16))) __bf16 v16bf;
typedef __attribute__((ext_vector_type(8)))  float  v8f;
typedef __attribute__((ext_vector_type(4)))  unsigned int u32x4;
typedef int v4i_vs __attribute__((vector_size(16)));   // matches builtin param type

union FragU { v16bf v; u32x4 q[2]; };

// float -> bf16 bits, round-to-nearest-even
__device__ __forceinline__ unsigned short f2bf(float f) {
    unsigned int u = __float_as_uint(f);
    unsigned int r = 0x7FFFu + ((u >> 16) & 1u);
    return (unsigned short)((u + r) >> 16);
}

// A-fragment (16x32, bf16) from row-major [*, lda] memory.
// lane l: row = mbase + (l&15); 16 contiguous K at kb..kb+15,
// kb = k0 + (l>>4)*16.
__device__ __forceinline__ v16bf load_frag_A(const unsigned short* __restrict__ base,
                                             int lda, int mbase, int k0, int lane) {
    int m  = mbase + (lane & 15);
    int kb = ((lane >> 4) << 4);
    const unsigned short* p = base + (long long)m * lda + kb + k0;
    FragU f;
    f.q[0] = *(const u32x4*)(p);
    f.q[1] = *(const u32x4*)(p + 8);
    return f.v;
}

// B-fragment (32x16, bf16) from N-major memory Bt[n][k] (row n = output col).
// lane l: n = nbase + (l&15); 16 contiguous K at k0 + (l>>4)*16.
__device__ __forceinline__ v16bf load_frag_B(const unsigned short* __restrict__ base,
                                             int ldb, int nbase, int k0, int lane) {
    int n  = nbase + (lane & 15);
    int ks = ((lane >> 4) << 4);
    const unsigned short* p = base + (long long)n * ldb + ks + k0;
    FragU f;
    f.q[0] = *(const u32x4*)(p);
    f.q[1] = *(const u32x4*)(p + 8);
    return f.v;
}

// Same, but clamps the row index (E rows past the edge only feed masked entries).
__device__ __forceinline__ v16bf load_frag_B_clamp(const unsigned short* __restrict__ base,
                                                   int ldb, int nbase, int k0, int lane,
                                                   int maxRow) {
    int n = nbase + (lane & 15);
    n = n < 0 ? 0 : (n > maxRow ? maxRow : n);
    int ks = k0 + ((lane >> 4) << 4);
    const unsigned short* p = base + (long long)n * ldb + ks;
    FragU f;
    f.q[0] = *(const u32x4*)(p);
    f.q[1] = *(const u32x4*)(p + 8);
    return f.v;
}

__device__ __forceinline__ v8f wmma_bf16(v16bf a, v16bf b, v8f c) {
    return __builtin_amdgcn_wmma_f32_16x16x32_bf16(false, a, false, b, (short)0, c,
                                                   false, false);
}

// ---- CDNA5 async global->LDS copy (ASYNCcnt-tracked), with safe fallback ----
__device__ __forceinline__ void async_copy16(const unsigned short* g, unsigned short* l) {
#if __has_builtin(__builtin_amdgcn_global_load_async_to_lds_b128)
    __builtin_amdgcn_global_load_async_to_lds_b128(
        (__attribute__((address_space(1))) v4i_vs*)g,
        (__attribute__((address_space(3))) v4i_vs*)l, 0, 0);
#else
    *(u32x4*)l = *(const u32x4*)g;
#endif
}

__device__ __forceinline__ void async_wait_all() {
#if __has_builtin(__builtin_amdgcn_s_wait_asynccnt)
    __builtin_amdgcn_s_wait_asynccnt(0);
#elif __has_builtin(__builtin_amdgcn_global_load_async_to_lds_b128)
    asm volatile("s_wait_asynccnt 0x0" ::: "memory");
#endif
}

// ---------------- kernel 0: conversions ----------------
__global__ __launch_bounds__(256)
void cvt_kernel(const float* __restrict__ x, const float* __restrict__ wqkv,
                const float* __restrict__ wproj, const float* __restrict__ E,
                unsigned short* __restrict__ xb, unsigned short* __restrict__ wqkvT,
                unsigned short* __restrict__ wprojT, unsigned short* __restrict__ Eb) {
    int job = blockIdx.y;
    long long tid = (long long)blockIdx.x * blockDim.x + threadIdx.x;
    long long stride = (long long)gridDim.x * blockDim.x;
    if (job == 0) {
        const long long N = (long long)S_LEN * HID;
        for (long long i = tid; i < N; i += stride) xb[i] = f2bf(x[i]);
    } else if (job == 1) {
        const long long N = (long long)H3 * HID;          // Wt[n][k] = W[k][n]
        for (long long i = tid; i < N; i += stride) {
            long long n = i / HID, k = i % HID;
            wqkvT[i] = f2bf(wqkv[k * H3 + n]);
        }
    } else if (job == 2) {
        const long long N = (long long)HID * HID;
        for (long long i = tid; i < N; i += stride) {
            long long n = i / HID, k = i % HID;
            wprojT[i] = f2bf(wproj[k * HID + n]);
        }
    } else {
        const long long N = (long long)NHEAD * S_LEN * DEPTH;
        for (long long i = tid; i < N; i += stride) Eb[i] = f2bf(E[i]);
    }
}

// ---------------- kernel 1: QKV GEMM (fully unrolled K loop) ----------------
__global__ __launch_bounds__(256)
void qkv_kernel(const unsigned short* __restrict__ xb,
                const unsigned short* __restrict__ wqkvT,
                const float* __restrict__ bias,
                unsigned short* __restrict__ Qb, unsigned short* __restrict__ Kb,
                unsigned short* __restrict__ Vt) {
    int wid = threadIdx.x >> 5, lane = threadIdx.x & 31;
    int w   = blockIdx.x * 8 + wid;
    int n64 = w % (H3 / 64);           // 0..47
    int mt  = w / (H3 / 64);           // 0..127
    int hl  = lane >> 4, n = lane & 15;

    v8f acc[4] = {};
#pragma unroll
    for (int kk = 0; kk < HID; kk += 32) {
        v16bf a = load_frag_A(xb, HID, mt * 16, kk, lane);
#pragma unroll
        for (int j = 0; j < 4; ++j) {
            v16bf b = load_frag_B(wqkvT, HID, n64 * 64 + j * 16, kk, lane);
            acc[j] = wmma_bf16(a, b, acc[j]);
        }
    }

    int part = (n64 * 64) / HID;       // 0=q 1=k 2=v
    int h    = ((n64 * 64) % HID) / DEPTH;
#pragma unroll
    for (int j = 0; j < 4; ++j) {
#pragma unroll
        for (int r = 0; r < 8; ++r) {
            int m  = mt * 16 + r + 8 * hl;
            int dd = j * 16 + n;
            float val = acc[j][r] + bias[n64 * 64 + dd];
            unsigned short bv = f2bf(val);
            if (part == 0)      Qb[((long long)h * S_LEN + m) * DEPTH + dd] = bv;
            else if (part == 1) Kb[((long long)h * S_LEN + m) * DEPTH + dd] = bv;
            else                Vt[((long long)h * DEPTH + dd) * S_LEN + m] = bv;
        }
    }
}

// ---------------- kernel 2: causal flash attention with rel bias ----------------
// grid (32 qblocks, 16 heads), 4 waves. K/V tiles staged in LDS via async copies,
// double-buffered so block kb+1 streams in while kb is computed.
__global__ __launch_bounds__(128)
void flash_kernel(const unsigned short* __restrict__ Qb,
                  const unsigned short* __restrict__ Kb,
                  const unsigned short* __restrict__ Vt,
                  const unsigned short* __restrict__ Eb,
                  unsigned short* __restrict__ Ctxb) {
    __shared__ __align__(16) unsigned short Ks[2][64 * 64];    // 16 KB
    __shared__ __align__(16) unsigned short Vs[2][64 * 64];    // 16 KB
    __shared__ __align__(16) unsigned short pbuf[4][16][64];   //  8 KB per-wave P
    int tid = threadIdx.x;
    int wid = tid >> 5, lane = tid & 31;
    int qb  = blockIdx.x, h = blockIdx.y;
    int q0  = qb * 64 + wid * 16;
    int hl  = lane >> 4, n = lane & 15;

    const unsigned short* Qh = Qb + (long long)h * S_LEN * DEPTH;
    const unsigned short* Kh = Kb + (long long)h * S_LEN * DEPTH;
    const unsigned short* Vh = Vt + (long long)h * DEPTH * S_LEN;
    const unsigned short* Eh = Eb + (long long)h * S_LEN * DEPTH;

    v16bf qf[2];
    qf[0] = load_frag_A(Qh, DEPTH, q0, 0, lane);
    qf[1] = load_frag_A(Qh, DEPTH, q0, 32, lane);

    v8f ctx[4] = {};
    float mrow[8], lrow[8];
    int sel[8], pidx[8];
#pragma unroll
    for (int r = 0; r < 8; ++r) {
        mrow[r] = -1e30f; lrow[r] = 0.0f;
        int M = r + 8 * hl;
        int coff = 15 - M + n;                 // rel source column offset, 0..30
        sel[r]  = (coff >= 16);
        pidx[r] = (((coff & 15) | (lane & 16)) << 2);   // ds_bpermute byte index
    }

    // stage K-tile (64x64 bf16, key-major) and V-tile (64x64 bf16, depth-major)
    auto stage = [&](int kblk, int buf) {
        int kk0 = kblk * 64;
#pragma unroll
        for (int i = 0; i < 4; ++i) {
            int id  = i * 128 + tid;           // 0..511 -> 512 x 16B = 8KB each
            int row = id >> 3;
            int ce  = (id & 7) << 3;
            async_copy16(Kh + (long long)(kk0 + row) * DEPTH + ce,
                         &Ks[buf][row * 64 + ce]);
            async_copy16(Vh + (long long)row * S_LEN + kk0 + ce,
                         &Vs[buf][row * 64 + ce]);
        }
    };

    stage(0, 0);
    for (int kb = 0; kb <= qb; ++kb) {
        int buf = kb & 1;
        int k0  = kb * 64;
        async_wait_all();          // this wave's staging for `buf` has landed
        __syncthreads();           // everyone's staging landed; prior reads done
        if (kb < qb) stage(kb + 1, buf ^ 1);   // overlap DMA with compute

        // --- rel strip R[ql, t] = q . E[mmin + t], t in [0,80) ---
        int mmin = (S_LEN - 1) - (q0 + 15) + k0;
        v8f R[5];
#pragma unroll
        for (int t = 0; t < 5; ++t) {
            v8f rt = {};
#pragma unroll
            for (int f = 0; f < 2; ++f) {
                v16bf e = load_frag_B_clamp(Eh, DEPTH, mmin + t * 16, f * 32, lane,
                                            S_LEN - 1);
                rt = wmma_bf16(qf[f], e, rt);
            }
            R[t] = rt;
        }
        // --- scores + skew-extracted rel + scale + causal mask ---
        float s[4][8];
#pragma unroll
        for (int j = 0; j < 4; ++j) {
            v8f sa = {};
#pragma unroll
            for (int f = 0; f < 2; ++f) {
                v16bf kf = load_frag_B(&Ks[buf][0], 64, j * 16, f * 32, lane);
                sa = wmma_bf16(qf[f], kf, sa);
            }
#pragma unroll
            for (int r = 0; r < 8; ++r) {
                float a = __int_as_float(
                    __builtin_amdgcn_ds_bpermute(pidx[r], __float_as_int(R[j][r])));
                float b = __int_as_float(
                    __builtin_amdgcn_ds_bpermute(pidx[r], __float_as_int(R[j + 1][r])));
                float rel = sel[r] ? b : a;
                float v = (sa[r] + rel) * 0.125f;      // rsqrt(64)
                if (kb == qb) {
                    int kg = k0 + j * 16 + n;
                    int qg = q0 + r + 8 * hl;
                    if (kg > qg) v = -10000.0f;        // matches reference mask
                }
                s[j][r] = v;
            }
        }
        // --- online softmax (per row = per (reg, lane-half)) ---
#pragma unroll
        for (int r = 0; r < 8; ++r) {
            float mx = fmaxf(fmaxf(s[0][r], s[1][r]), fmaxf(s[2][r], s[3][r]));
#pragma unroll
            for (int off = 1; off < 16; off <<= 1)
                mx = fmaxf(mx, __shfl_xor(mx, off, 32));
            float mnew = fmaxf(mrow[r], mx);
            float p0 = __expf(s[0][r] - mnew), p1 = __expf(s[1][r] - mnew);
            float p2 = __expf(s[2][r] - mnew), p3 = __expf(s[3][r] - mnew);
            float ssum = p0 + p1 + p2 + p3;
#pragma unroll
            for (int off = 1; off < 16; off <<= 1)
                ssum += __shfl_xor(ssum, off, 32);
            float alpha = __expf(mrow[r] - mnew);
            mrow[r] = mnew;
            lrow[r] = lrow[r] * alpha + ssum;
#pragma unroll
            for (int jd = 0; jd < 4; ++jd) ctx[jd][r] *= alpha;
            int M = r + 8 * hl;
            pbuf[wid][M][0 * 16 + n] = f2bf(p0);
            pbuf[wid][M][1 * 16 + n] = f2bf(p1);
            pbuf[wid][M][2 * 16 + n] = f2bf(p2);
            pbuf[wid][M][3 * 16 + n] = f2bf(p3);
        }
        // --- PV: refragment P from per-wave LDS, multiply into ctx ---
        v16bf pf[2];
        pf[0] = load_frag_A(&pbuf[wid][0][0], 64, 0, 0, lane);
        pf[1] = load_frag_A(&pbuf[wid][0][0], 64, 0, 32, lane);
#pragma unroll
        for (int jd = 0; jd < 4; ++jd) {
#pragma unroll
            for (int f = 0; f < 2; ++f) {
                v16bf vf = load_frag_B(&Vs[buf][0], 64, jd * 16, f * 32, lane);
                ctx[jd] = wmma_bf16(pf[f], vf, ctx[jd]);
            }
        }
    }
    // --- finalize: ctx / l, merge heads, bf16 for projection GEMM ---
#pragma unroll
    for (int jd = 0; jd < 4; ++jd) {
#pragma unroll
        for (int r = 0; r < 8; ++r) {
            int m   = q0 + r + 8 * hl;
            int col = h * DEPTH + jd * 16 + n;
            Ctxb[(long long)m * HID + col] = f2bf(ctx[jd][r] / lrow[r]);
        }
    }
}

// ---------------- kernel 3: output projection (fully unrolled K loop) ----------------
__global__ __launch_bounds__(256)
void proj_kernel(const unsigned short* __restrict__ Ctxb,
                 const unsigned short* __restrict__ wprojT,
                 const float* __restrict__ bias, float* __restrict__ out) {
    int wid = threadIdx.x >> 5, lane = threadIdx.x & 31;
    int w   = blockIdx.x * 8 + wid;
    int n64 = w % (HID / 64);          // 0..15
    int mt  = w / (HID / 64);          // 0..127
    int hl  = lane >> 4, n = lane & 15;

    v8f acc[4] = {};
#pragma unroll
    for (int kk = 0; kk < HID; kk += 32) {
        v16bf a = load_frag_A(Ctxb, HID, mt * 16, kk, lane);
#pragma unroll
        for (int j = 0; j < 4; ++j) {
            v16bf b = load_frag_B(wprojT, HID, n64 * 64 + j * 16, kk, lane);
            acc[j] = wmma_bf16(a, b, acc[j]);
        }
    }

#pragma unroll
    for (int j = 0; j < 4; ++j) {
#pragma unroll
        for (int r = 0; r < 8; ++r) {
            int m     = mt * 16 + r + 8 * hl;
            int nglob = n64 * 64 + j * 16 + n;
            out[(long long)m * HID + nglob] = acc[j][r] + bias[nglob];
        }
    }
}

// ---------------- host launch ----------------
extern "C" void kernel_launch(void* const* d_in, const int* in_sizes, int n_in,
                              void* d_out, int out_size, void* d_ws, size_t ws_size,
                              hipStream_t stream) {
    const float* x        = (const float*)d_in[0];
    const float* c_attn_w = (const float*)d_in[1];
    const float* c_attn_b = (const float*)d_in[2];
    const float* c_proj_w = (const float*)d_in[3];
    const float* c_proj_b = (const float*)d_in[4];
    const float* E        = (const float*)d_in[5];
    float* out = (float*)d_out;

    char* ws = (char*)d_ws;
    size_t off = 0;
    auto take = [&](size_t bytes) -> char* {
        char* p = ws + off;
        off = (off + bytes + 255) & ~(size_t)255;
        return p;
    };
    unsigned short* xb     = (unsigned short*)take((size_t)S_LEN * HID * 2);
    unsigned short* wqkvT  = (unsigned short*)take((size_t)H3 * HID * 2);
    unsigned short* wprojT = (unsigned short*)take((size_t)HID * HID * 2);
    unsigned short* Eb     = (unsigned short*)take((size_t)NHEAD * S_LEN * DEPTH * 2);
    unsigned short* Qb     = (unsigned short*)take((size_t)NHEAD * S_LEN * DEPTH * 2);
    unsigned short* Kb     = (unsigned short*)take((size_t)NHEAD * S_LEN * DEPTH * 2);
    unsigned short* Vt     = (unsigned short*)take((size_t)NHEAD * S_LEN * DEPTH * 2);
    unsigned short* Ctxb   = (unsigned short*)take((size_t)S_LEN * HID * 2);
    (void)in_sizes; (void)n_in; (void)out_size; (void)ws_size;

    cvt_kernel<<<dim3(1024, 4), 256, 0, stream>>>(x, c_attn_w, c_proj_w, E,
                                                  xb, wqkvT, wprojT, Eb);
    qkv_kernel<<<dim3(768), 256, 0, stream>>>(xb, wqkvT, c_attn_b, Qb, Kb, Vt);
    flash_kernel<<<dim3(32, 16), 128, 0, stream>>>(Qb, Kb, Vt, Eb, Ctxb);
    proj_kernel<<<dim3(256), 256, 0, stream>>>(Ctxb, wprojT, c_proj_b, out);
}
